// AttentiveEncoder_78597901517476
// MI455X (gfx1250) — compile-verified, matrix operand order checked by reading, not run on hardware
//
#include <hip/hip_runtime.h>
#include <hip/hip_bf16.h>

typedef __attribute__((ext_vector_type(16))) __bf16          v16bf;
typedef __attribute__((ext_vector_type(8)))  float           v8f;
typedef __attribute__((ext_vector_type(8)))  unsigned short  u16x8;
typedef __attribute__((ext_vector_type(4)))  float           f32x4;

struct BfPack { u16x8 lo, hi; };

static __device__ __forceinline__ v16bf pack_bf(u16x8 lo, u16x8 hi) {
    BfPack p; p.lo = lo; p.hi = hi;
    return __builtin_bit_cast(v16bf, p);
}

static __device__ __forceinline__ unsigned short f2bf(float f) {
    return __builtin_bit_cast(unsigned short, (__bf16)f);
}

#define SEQ 8192
#define HID 1024

// -------------------------------------------------------------------------
// Kernel 1: lin = bf16( emb_table[ids] @ W^T + b ), also store lin^T.
// One wave per 16x16 output tile; K=1024 in 32 steps of wmma 16x16x32 bf16.
// -------------------------------------------------------------------------
__global__ void k_embed_linear(const int* __restrict__ ids,
                               const float* __restrict__ emb,
                               const float* __restrict__ W,
                               const float* __restrict__ bias,
                               unsigned short* __restrict__ lin,    // [SEQ][HID]
                               unsigned short* __restrict__ linT)   // [HID][SEQ]
{
    const int wv   = threadIdx.x >> 5;
    const int lane = threadIdx.x & 31;
    const int col  = lane & 15;       // A row / B col this lane serves
    const int hi   = lane >> 4;       // lane group (0:lanes 0-15, 1:lanes 16-31)
    const int b8   = hi * 8;          // K sub-chunk base per 16-bit WMMA layout

    const int tile = blockIdx.x * 4 + wv;     // 32768 tiles total
    const int mt   = tile >> 6;               // 512 row tiles
    const int nt   = tile & 63;               // 64  col tiles

    const int   arow = mt * 16 + col;
    const int   id   = ids[arow];
    const float* ap  = emb + (size_t)id * HID;
    const float* bp  = W   + (size_t)(nt * 16 + col) * HID;

    v8f acc = {};
    for (int kc = 0; kc < 32; ++kc) {
        const int k0 = kc * 32 + b8;
        f32x4 a0 = *(const f32x4*)(ap + k0);
        f32x4 a1 = *(const f32x4*)(ap + k0 + 4);
        f32x4 a2 = *(const f32x4*)(ap + k0 + 16);
        f32x4 a3 = *(const f32x4*)(ap + k0 + 20);
        f32x4 b0 = *(const f32x4*)(bp + k0);
        f32x4 b1 = *(const f32x4*)(bp + k0 + 4);
        f32x4 b2 = *(const f32x4*)(bp + k0 + 16);
        f32x4 b3 = *(const f32x4*)(bp + k0 + 20);
        v16bf A, B;
#pragma unroll
        for (int t = 0; t < 4; ++t) {
            A[t]      = (__bf16)a0[t];  A[4 + t]  = (__bf16)a1[t];
            A[8 + t]  = (__bf16)a2[t];  A[12 + t] = (__bf16)a3[t];
            B[t]      = (__bf16)b0[t];  B[4 + t]  = (__bf16)b1[t];
            B[8 + t]  = (__bf16)b2[t];  B[12 + t] = (__bf16)b3[t];
        }
        acc = __builtin_amdgcn_wmma_f32_16x16x32_bf16(false, A, false, B,
                                                      (short)0, acc, false, false);
    }

    const float bb = bias[nt * 16 + col];
    u16x8 tcol;
#pragma unroll
    for (int i = 0; i < 8; ++i) {
        const float v = acc[i] + bb;
        const unsigned short u = f2bf(v);
        const int r = mt * 16 + hi * 8 + i;
        lin[(size_t)r * HID + nt * 16 + col] = u;   // row-major (Q, K)
        tcol[i] = u;
    }
    *(u16x8*)(linT + (size_t)(nt * 16 + col) * SEQ + mt * 16 + hi * 8) = tcol;
}

// -------------------------------------------------------------------------
// Kernel 2: flash attention, one block per 32-query tile (2 q-subtiles),
// 16 waves (512 thr). Per 128-key iteration:
//   wave w computes S subtile (qt = w>>3, kt = w&7) -> online softmax across
//   waves via LDS -> P (bf16) in LDS -> wave w accumulates O for
//   rows qt*16..+15, cols (w&7)*128..+127 (8 h-tiles, 64 acc VGPRs).
// Q tile (64 KB) staged via GLOBAL_LOAD_ASYNC_TO_LDS_B128.
// -------------------------------------------------------------------------
__global__ void __launch_bounds__(512)
k_flash_attn(const unsigned short* __restrict__ lin,   // [SEQ][HID] bf16 (Q,K)
             const unsigned short* __restrict__ linT,  // [HID][SEQ] bf16 (V^T)
             float* __restrict__ out)                  // [SEQ][HID] f32
{
    __shared__ unsigned short Qs[32 * HID];   // 64 KB: query tile (32 rows)
    __shared__ unsigned short Ps[32 * 128];   //  8 KB: exp'd probs, bf16
    __shared__ float red[32 * 8];             // [row][key-subtile] max / sum
    __shared__ float m_sh[32], l_sh[32], newm_sh[32], resc_sh[32];

    const int tid  = threadIdx.x;
    const int wv   = tid >> 5;
    const int lane = tid & 31;
    const int col  = lane & 15;
    const int hi   = lane >> 4;
    const int b8   = hi * 8;
    const int q    = blockIdx.x;              // 32-query block index (256)

    const int qt    = wv >> 3;                // which 16-row q-subtile (0..1)
    const int kt    = wv & 7;                 // which 16-key subtile   (0..7)
    const int hbase = (wv & 7) * 128;         // owned output columns

    // ---- stage 32x1024 bf16 Q tile into LDS with async DMA (ASYNCcnt) ----
    {
        const char*    gsrc  = (const char*)(lin + (size_t)q * 32 * HID);
        const unsigned lbase = (unsigned)(size_t)(void*)Qs;   // LDS byte offset
#pragma unroll
        for (int t = 0; t < 8; ++t) {
            const unsigned           boff = (unsigned)(tid + 512 * t) * 16u;
            const unsigned           ldsa = lbase + boff;
            const unsigned long long ga   = (unsigned long long)(gsrc + boff);
            asm volatile("global_load_async_to_lds_b128 %0, %1, off"
                         :: "v"(ldsa), "v"(ga) : "memory");
        }
        asm volatile("s_wait_asynccnt 0" ::: "memory");
    }
    if (tid < 32) { m_sh[tid] = -3.0e38f; l_sh[tid] = 0.0f; }
    __syncthreads();

    v8f o[8] = {};                            // 16 rows x 128 cols f32
    const float scale = 0.03125f;             // 1/sqrt(1024)

    for (int kb = 0; kb < 64; ++kb) {
        const int j0 = kb * 128;

        // ---- S = (Q @ K^T) * scale : 32 WMMAs over K=1024 ----
        v8f s = {};
        const unsigned short* kp = lin + (size_t)(j0 + kt * 16 + col) * HID;
        const unsigned short* qp = Qs + (qt * 16 + col) * HID;
        if (kb + 1 < 64)   // prefetch next key tile (global_prefetch_b8)
            __builtin_prefetch(lin + (size_t)(j0 + 128 + kt * 16 + col) * HID, 0, 1);
        for (int kc = 0; kc < 32; ++kc) {
            const int k0 = kc * 32 + b8;
            v16bf A = pack_bf(*(const u16x8*)(qp + k0), *(const u16x8*)(qp + k0 + 16));
            v16bf B = pack_bf(*(const u16x8*)(kp + k0), *(const u16x8*)(kp + k0 + 16));
            s = __builtin_amdgcn_wmma_f32_16x16x32_bf16(false, A, false, B,
                                                        (short)0, s, false, false);
        }
#pragma unroll
        for (int i = 0; i < 8; ++i) s[i] *= scale;

        // ---- per-row max of this wave's subtile ----
        float rm[8];
#pragma unroll
        for (int i = 0; i < 8; ++i) rm[i] = s[i];
#pragma unroll
        for (int off = 8; off >= 1; off >>= 1)
#pragma unroll
            for (int i = 0; i < 8; ++i) rm[i] = fmaxf(rm[i], __shfl_xor(rm[i], off, 32));
        if (lane == 0)  { for (int i = 0; i < 8; ++i) red[(qt * 16 + i) * 8 + kt]     = rm[i]; }
        if (lane == 16) { for (int i = 0; i < 8; ++i) red[(qt * 16 + 8 + i) * 8 + kt] = rm[i]; }
        __syncthreads();

        // ---- stats waves 0/1: new running max + rescale per query row ----
        if (wv < 2 && lane < 16) {
            const int r  = wv * 16 + lane;
            float mo = m_sh[r], mm = mo;
            for (int k = 0; k < 8; ++k) mm = fmaxf(mm, red[r * 8 + k]);
            newm_sh[r] = mm;
            resc_sh[r] = __expf(mo - mm);
            m_sh[r]    = mm;
        }
        __syncthreads();

        // ---- P = exp(S - m) -> LDS bf16; row sums; rescale O ----
        float rs[8];
#pragma unroll
        for (int i = 0; i < 8; ++i) {
            const int   r = qt * 16 + hi * 8 + i;
            const float p = __expf(s[i] - newm_sh[r]);
            rs[i] = p;
            Ps[r * 128 + kt * 16 + col] = f2bf(p);
        }
#pragma unroll
        for (int off = 8; off >= 1; off >>= 1)
#pragma unroll
            for (int i = 0; i < 8; ++i) rs[i] += __shfl_xor(rs[i], off, 32);
        if (lane == 0)  { for (int i = 0; i < 8; ++i) red[(qt * 16 + i) * 8 + kt]     = rs[i]; }
        if (lane == 16) { for (int i = 0; i < 8; ++i) red[(qt * 16 + 8 + i) * 8 + kt] = rs[i]; }
#pragma unroll
        for (int i = 0; i < 8; ++i) {
            const float r = resc_sh[qt * 16 + hi * 8 + i];
#pragma unroll
            for (int ht = 0; ht < 8; ++ht) o[ht][i] *= r;
        }
        __syncthreads();

        // ---- stats waves: l = l * rescale + sum(exp) ----
        if (wv < 2 && lane < 16) {
            const int r = wv * 16 + lane;
            float tot = 0.0f;
            for (int k = 0; k < 8; ++k) tot += red[r * 8 + k];
            l_sh[r] = l_sh[r] * resc_sh[r] + tot;
        }

        // ---- O += P @ V : K=128 in 4 steps, 8 h-tiles per wave ----
        const unsigned short* pp = Ps + (qt * 16 + col) * 128;
        for (int kc = 0; kc < 4; ++kc) {
            const int kk = kc * 32 + b8;
            v16bf A = pack_bf(*(const u16x8*)(pp + kk), *(const u16x8*)(pp + kk + 16));
            const size_t jb = (size_t)j0 + kc * 32 + b8;
#pragma unroll
            for (int ht = 0; ht < 8; ++ht) {
                const unsigned short* vp = linT + (size_t)(hbase + ht * 16 + col) * SEQ + jb;
                v16bf B = pack_bf(*(const u16x8*)vp, *(const u16x8*)(vp + 16));
                o[ht] = __builtin_amdgcn_wmma_f32_16x16x32_bf16(false, A, false, B,
                                                                (short)0, o[ht], false, false);
            }
        }
        __syncthreads();   // protect red/Ps/newm/resc for next iter; publishes l
    }

    // ---- finalize: divide by l, store f32 output ----
#pragma unroll
    for (int i = 0; i < 8; ++i) {
        const int   r    = qt * 16 + hi * 8 + i;
        const float invl = 1.0f / l_sh[r];
        float* op = out + (size_t)(q * 32 + r) * HID + hbase + col;
#pragma unroll
        for (int ht = 0; ht < 8; ++ht) op[ht * 16] = o[ht][i] * invl;
    }
}

// -------------------------------------------------------------------------
extern "C" void kernel_launch(void* const* d_in, const int* in_sizes, int n_in,
                              void* d_out, int out_size, void* d_ws, size_t ws_size,
                              hipStream_t stream) {
    const int*   ids  = (const int*)d_in[0];
    const float* emb  = (const float*)d_in[1];
    const float* W    = (const float*)d_in[2];
    const float* bias = (const float*)d_in[3];
    float*       out  = (float*)d_out;

    unsigned short* lin  = (unsigned short*)d_ws;                    // 16 MB bf16 [SEQ][HID]
    unsigned short* linT = lin + (size_t)SEQ * HID;                  // 16 MB bf16 [HID][SEQ]

    k_embed_linear<<<8192, 128, 0, stream>>>(ids, emb, W, bias, lin, linT);
    k_flash_attn<<<256, 512, 0, stream>>>(lin, linT, out);
}
